// RotarySelfAttention_15092515078775
// MI455X (gfx1250) — matrix operand order repaired and checked by reading, not compiled, not run on hardware
//
#include <hip/hip_runtime.h>

// Problem constants (from reference)
#define BB   2
#define SS   2048
#define DD   768
#define HH   12
#define HDD  64
#define EQKV 2304   // 3*DD

typedef __attribute__((ext_vector_type(16))) __bf16       bf16x16;
typedef __attribute__((ext_vector_type(8)))  float        f32x8;
typedef __attribute__((ext_vector_type(4)))  float        f32x4;
typedef __attribute__((ext_vector_type(4)))  unsigned int u32x4;
typedef __attribute__((ext_vector_type(4)))  int          i32x4;

// gfx1250 async global->LDS path (ASYNCcnt), with sync fallback.
// Builtin signature (from probe): (v4i __device__*, v4i __shared__*, imm, imm)
#if defined(__AMDGCN__) && __has_builtin(__builtin_amdgcn_global_load_async_to_lds_b128) && __has_builtin(__builtin_amdgcn_s_wait_asynccnt)
#define ASYNC_OK 1
typedef __attribute__((address_space(1))) i32x4 GV4;   // global v4i
typedef __attribute__((address_space(3))) i32x4 LV4;   // LDS v4i
#else
#define ASYNC_OK 0
#endif

union Frag {            // one WMMA 16-bit A or B fragment (16 bf16 / 8 VGPRs)
  bf16x16 v;
  u32x4   q[2];
  __bf16  e[16];
};
union Pack8 {           // 8 bf16 packed for one 16-byte store
  u32x4  q;
  __bf16 e[8];
};

static __device__ __forceinline__ f32x8 wmma_bf16(const Frag& a, const Frag& b, f32x8 c) {
  return __builtin_amdgcn_wmma_f32_16x16x32_bf16(false, a.v, false, b.v, (short)0, c, false, false);
}

static __device__ __forceinline__ float red16_max(float v) {
  v = fmaxf(v, __shfl_xor(v, 1));
  v = fmaxf(v, __shfl_xor(v, 2));
  v = fmaxf(v, __shfl_xor(v, 4));
  v = fmaxf(v, __shfl_xor(v, 8));
  return v;   // masks 1,2,4,8 stay inside each 16-lane half (rows differ per half)
}
static __device__ __forceinline__ float red16_sum(float v) {
  v += __shfl_xor(v, 1);
  v += __shfl_xor(v, 2);
  v += __shfl_xor(v, 4);
  v += __shfl_xor(v, 8);
  return v;
}

// -------------------------------------------------------------------------
// Kernel 1: qkv = w_qkv (2304x768) * x^T (768 x S) computed as C[e,s] tiles,
// RoPE applied in-register to q/k rows, outputs written as:
//   Q,K : bf16 [B,H,S,64]   (hd contiguous; Q pre-scaled by 1/8)
//   Vt  : bf16 [B,H,64,S]   (s contiguous, for PV B-fragments)
// -------------------------------------------------------------------------
__global__ __launch_bounds__(256) void qkv_rope_kernel(
    const float* __restrict__ x, const float* __restrict__ w_qkv,
    __bf16* __restrict__ Qb, __bf16* __restrict__ Kb, __bf16* __restrict__ Vt)
{
  const int lane = threadIdx.x & 31;
  const int wave = threadIdx.x >> 5;
  int tile = blockIdx.x * 8 + wave;
  const int TPB = (EQKV / 16) * (SS / 16);          // tiles per batch
  const int b   = tile / TPB;  tile %= TPB;
  const int et  = tile / (SS / 16);
  const int st  = tile % (SS / 16);
  const int e0  = et * 16, s0 = st * 16;
  const int  ln = lane & 15;
  const bool hi = lane >= 16;

  const float* __restrict__ wrow = w_qkv + (size_t)(e0 + ln) * DD;        // A: row e
  const float* __restrict__ xrow = x + ((size_t)b * SS + (s0 + ln)) * DD; // B: col s

  f32x8 acc = {};
  for (int d0 = 0; d0 < DD; d0 += 32) {
    __builtin_prefetch(xrow + d0 + 64, 0, 0);
    Frag a, bf;
    const int ka = d0 + (hi ? 16 : 0);      // A: lanes 0-15 -> K 0..15 ; 16-31 -> 16..31
#pragma unroll
    for (int j = 0; j < 8; ++j) { a.e[j] = (__bf16)wrow[ka + j]; a.e[8 + j] = (__bf16)wrow[ka + 8 + j]; }
    const int kb = d0 + (hi ? 16 : 0);      // B: lanes 0-15 -> K 0..15 ; 16-31 -> 16..31
#pragma unroll
    for (int j = 0; j < 16; ++j) bf.e[j] = (__bf16)xrow[kb + j];
    acc = wmma_bf16(a, bf, acc);
  }

  // D layout: lane (0-15)=col s0+ln ; VGPR r = row e0+r (lo half) / e0+8+r (hi half)
  const int region  = e0 / DD;              // 0=q 1=k 2=v
  const int h       = (e0 % DD) / HDD;
  const int hd0     = e0 % HDD;             // 0,16,32,48
  const int s       = s0 + ln;
  const int rowbase = hd0 + (hi ? 8 : 0);   // hd of accumulator r=0 (always even)

  if (region < 2) {                         // RoPE on q and k (interleaved pairs)
#pragma unroll
    for (int j = 0; j < 4; ++j) {
      const int hd_even = rowbase + 2 * j;
      const float invf  = exp2f(-0.20762050593046f * (float)hd_even); // 10000^(-hd/64)
      const float ang   = (float)s * invf;
      float sn, cs;
      __sincosf(ang, &sn, &cs);
      const float x0 = acc[2 * j], x1 = acc[2 * j + 1];
      acc[2 * j]     = x0 * cs - x1 * sn;
      acc[2 * j + 1] = x1 * cs + x0 * sn;
    }
  }

  if (region < 2) {
    const float sc = (region == 0) ? 0.125f : 1.0f;   // fold 1/sqrt(HD) into Q
    Pack8 p;
#pragma unroll
    for (int r = 0; r < 8; ++r) p.e[r] = (__bf16)(acc[r] * sc);
    __bf16* dst = (region == 0 ? Qb : Kb);
    *(u32x4*)(dst + (((size_t)b * HH + h) * SS + s) * HDD + rowbase) = p.q;
  } else {                                  // V -> transposed [B,H,64,S]
#pragma unroll
    for (int r = 0; r < 8; ++r)
      Vt[(((size_t)b * HH + h) * HDD + (rowbase + r)) * SS + s] = (__bf16)acc[r];
  }
}

// -------------------------------------------------------------------------
// Kernel 2: flash attention per (b,h). Block = 8 waves, each wave owns 16 q
// rows. K/V 32-key chunks are staged once per block into LDS, double-buffered
// with GLOBAL_LOAD_ASYNC_TO_LDS (ASYNCcnt) when available; fragments are read
// back with ds_load_b128. Online softmax; output bf16 [B,S,768].
// -------------------------------------------------------------------------
__global__ __launch_bounds__(256) void attn_kernel(
    const __bf16* __restrict__ Qb, const __bf16* __restrict__ Kb,
    const __bf16* __restrict__ Vt, __bf16* __restrict__ Ob)
{
  __shared__ __bf16 kbuf[2][32][64];        // 2 x 4KB : K chunk, [krow][hd]
  __shared__ __bf16 vbuf[2][64][32];        // 2 x 4KB : V chunk, [hd][krow]
  __shared__ __bf16 lds_p[8][16][32];       // per-wave 16x32 P tile for transpose

  const int tid  = threadIdx.x;
  const int lane = tid & 31;
  const int wave = tid >> 5;
  const int bh   = blockIdx.x >> 4;         // 0..23
  const int qt   = blockIdx.x & 15;
  const int b    = bh / HH, h = bh % HH;
  const int q0   = qt * 128 + wave * 16;
  const int  ln  = lane & 15;
  const bool hi  = lane >= 16;

  const __bf16* __restrict__ kbase = Kb + (size_t)bh * SS * HDD;
  const __bf16* __restrict__ vbase = Vt + (size_t)bh * HDD * SS;

  // Cooperative chunk staging: 256 threads x 16B cover each 4KB buffer.
  auto issue_prefetch = [&](int kc, int bufi) {
    const __bf16* gk = kbase + (size_t)(kc + (tid >> 3)) * HDD + (tid & 7) * 8;
    const __bf16* gv = vbase + (size_t)(tid >> 2) * SS + kc + (tid & 3) * 8;
    __bf16* lk = &kbuf[bufi][tid >> 3][(tid & 7) * 8];
    __bf16* lv = &vbuf[bufi][tid >> 2][(tid & 3) * 8];
#if ASYNC_OK
    __builtin_amdgcn_global_load_async_to_lds_b128((GV4*)gk, (LV4*)lk, 0, 0);
    __builtin_amdgcn_global_load_async_to_lds_b128((GV4*)gv, (LV4*)lv, 0, 0);
#else
    *(u32x4*)lk = *(const u32x4*)gk;
    *(u32x4*)lv = *(const u32x4*)gv;
#endif
  };

  // Q A-fragments for hd windows [0,32) and [32,64): resident for whole loop
  const __bf16* __restrict__ qptr = Qb + (((size_t)bh) * SS + q0 + ln) * HDD;
  const int ka = hi ? 16 : 0;
  Frag aq0, aq1;
  aq0.q[0] = *(const u32x4*)(qptr + ka);
  aq0.q[1] = *(const u32x4*)(qptr + ka + 8);
  aq1.q[0] = *(const u32x4*)(qptr + 32 + ka);
  aq1.q[1] = *(const u32x4*)(qptr + 32 + ka + 8);

  f32x8 o[4] = {f32x8{}, f32x8{}, f32x8{}, f32x8{}};
  float m[8], l[8];
#pragma unroll
  for (int r = 0; r < 8; ++r) { m[r] = -1e30f; l[r] = 0.0f; }

  issue_prefetch(0, 0);

  for (int i = 0; i < SS / 32; ++i) {
    const int cur = i & 1;
#if ASYNC_OK
    __builtin_amdgcn_s_wait_asynccnt(0);    // own async fills landed
#endif
    __syncthreads();                        // everyone's fills visible; prev buffer free
    if (i + 1 < SS / 32) issue_prefetch((i + 1) * 32, 1 - cur);

    // ---- scores: two 16-col tiles (local k rows 0..15 / 16..31), K=64 in 2 WMMAs each
    const __bf16* kr0 = &kbuf[cur][ln][hi ? 16 : 0];
    const __bf16* kr1 = &kbuf[cur][16 + ln][hi ? 16 : 0];
    Frag b00, b01, b10, b11;
    b00.q[0] = *(const u32x4*)(kr0);        b00.q[1] = *(const u32x4*)(kr0 + 8);
    b01.q[0] = *(const u32x4*)(kr0 + 32);   b01.q[1] = *(const u32x4*)(kr0 + 40);
    b10.q[0] = *(const u32x4*)(kr1);        b10.q[1] = *(const u32x4*)(kr1 + 8);
    b11.q[0] = *(const u32x4*)(kr1 + 32);   b11.q[1] = *(const u32x4*)(kr1 + 40);
    f32x8 sA = {}, sB = {};
    sA = wmma_bf16(aq0, b00, sA);  sA = wmma_bf16(aq1, b01, sA);
    sB = wmma_bf16(aq0, b10, sB);  sB = wmma_bf16(aq1, b11, sB);

    // ---- online softmax (rows in VGPR index r, cols across 16-lane group)
    float pA[8], pB[8];
#pragma unroll
    for (int r = 0; r < 8; ++r) {
      const float cmax = red16_max(fmaxf(sA[r], sB[r]));
      const float mn   = fmaxf(m[r], cmax);
      const float sc   = __expf(m[r] - mn);
      m[r] = mn;
      pA[r] = __expf(sA[r] - mn);
      pB[r] = __expf(sB[r] - mn);
      l[r]  = l[r] * sc + red16_sum(pA[r] + pB[r]);
      o[0][r] *= sc; o[1][r] *= sc; o[2][r] *= sc; o[3][r] *= sc;
    }

    // ---- transpose P (D-layout -> A-layout) through per-wave LDS region.
    // Same-wave LDS ops execute in order; region is wave-private, no barrier.
#pragma unroll
    for (int r = 0; r < 8; ++r) {
      lds_p[wave][(hi ? 8 : 0) + r][ln]      = (__bf16)pA[r];
      lds_p[wave][(hi ? 8 : 0) + r][16 + ln] = (__bf16)pB[r];
    }
    Frag ap;
    const __bf16* prow = &lds_p[wave][ln][0];
    ap.q[0] = *(const u32x4*)(prow + ka);        // K 0..7  / 16..23
    ap.q[1] = *(const u32x4*)(prow + 8 + ka);    // K 8..15 / 24..31

    // ---- O += P * V  (4 hd tiles of 16), V B-frags from LDS
#pragma unroll
    for (int t = 0; t < 4; ++t) {
      const __bf16* vr = &vbuf[cur][t * 16 + ln][hi ? 16 : 0];
      Frag bv;
      bv.q[0] = *(const u32x4*)(vr);
      bv.q[1] = *(const u32x4*)(vr + 8);
      o[t] = wmma_bf16(ap, bv, o[t]);
    }
  }

  // ---- finalize: O/l  -> attn buffer [B,S,768] bf16
  const size_t obase = ((size_t)b * SS + q0) * DD + (size_t)h * HDD;
#pragma unroll
  for (int r = 0; r < 8; ++r) {
    const float inv = 1.0f / l[r];
    const int  row  = (hi ? 8 : 0) + r;
#pragma unroll
    for (int t = 0; t < 4; ++t)
      Ob[obase + (size_t)row * DD + t * 16 + ln] = (__bf16)(o[t][r] * inv);
  }
}

// -------------------------------------------------------------------------
// Kernel 3: out = attn (B,S,768) * w_out^T + b_out, computed as C[e,s] tiles.
// -------------------------------------------------------------------------
__global__ __launch_bounds__(256) void out_proj_kernel(
    const __bf16* __restrict__ attn, const float* __restrict__ w_out,
    const float* __restrict__ b_out, float* __restrict__ out)
{
  const int lane = threadIdx.x & 31;
  const int wave = threadIdx.x >> 5;
  int tile = blockIdx.x * 8 + wave;
  const int TPB = (DD / 16) * (SS / 16);
  const int b   = tile / TPB;  tile %= TPB;
  const int et  = tile / (SS / 16);
  const int st  = tile % (SS / 16);
  const int e0  = et * 16, s0 = st * 16;
  const int  ln = lane & 15;
  const bool hi = lane >= 16;

  const float*  __restrict__ wrow = w_out + (size_t)(e0 + ln) * DD;
  const __bf16* __restrict__ arow = attn + ((size_t)b * SS + s0 + ln) * DD;

  f32x8 acc = {};
  for (int d0 = 0; d0 < DD; d0 += 32) {
    __builtin_prefetch(arow + d0 + 64, 0, 0);
    Frag a, bf;
    const int kaw = d0 + (hi ? 16 : 0);
#pragma unroll
    for (int j = 0; j < 8; ++j) { a.e[j] = (__bf16)wrow[kaw + j]; a.e[8 + j] = (__bf16)wrow[kaw + 8 + j]; }
    const int kb = d0 + (hi ? 16 : 0);
    bf.q[0] = *(const u32x4*)(arow + kb);
    bf.q[1] = *(const u32x4*)(arow + kb + 8);
    acc = wmma_bf16(a, bf, acc);
  }

  const int ebase = e0 + (hi ? 8 : 0);
  float* orow = out + ((size_t)b * SS + s0 + ln) * DD + ebase;
  f32x4 lo, hiv;
#pragma unroll
  for (int r = 0; r < 4; ++r) { lo[r] = acc[r] + b_out[ebase + r]; hiv[r] = acc[4 + r] + b_out[ebase + 4 + r]; }
  *(f32x4*)(orow)     = lo;
  *(f32x4*)(orow + 4) = hiv;
}

// -------------------------------------------------------------------------
extern "C" void kernel_launch(void* const* d_in, const int* in_sizes, int n_in,
                              void* d_out, int out_size, void* d_ws, size_t ws_size,
                              hipStream_t stream) {
  const float* x     = (const float*)d_in[0];
  const float* w_qkv = (const float*)d_in[1];
  const float* w_out = (const float*)d_in[2];
  const float* b_out = (const float*)d_in[3];
  float* out = (float*)d_out;

  const size_t nQ = (size_t)BB * HH * SS * HDD;     // 3,145,728 elems
  __bf16* Qb   = (__bf16*)d_ws;                     // 6 MB
  __bf16* Kb   = Qb + nQ;                           // 6 MB
  __bf16* Vt   = Kb + nQ;                           // 6 MB (transposed V)
  __bf16* Attn = Vt + nQ;                           // 6 MB [B,S,768]

  qkv_rope_kernel<<<(BB * (EQKV / 16) * (SS / 16)) / 8, 256, 0, stream>>>(x, w_qkv, Qb, Kb, Vt);
  attn_kernel<<<BB * HH * (SS / 128), 256, 0, stream>>>(Qb, Kb, Vt, Attn);
  out_proj_kernel<<<(BB * (DD / 16) * (SS / 16)) / 8, 256, 0, stream>>>(Attn, w_out, b_out, out);
}
